// SetAbstraction_3762391351571
// MI455X (gfx1250) — matrix operand order repaired and checked by reading, not compiled, not run on hardware
//
#include <hip/hip_runtime.h>

#define B_   8
#define N_   32768
#define C_   64
#define O_   128
#define S_   1024
#define R2_  0.01f
#define NEGF (-3.402823466e38f)

typedef __attribute__((ext_vector_type(16))) __bf16       v16bf;
typedef __attribute__((ext_vector_type(8)))  float        v8f;
typedef __attribute__((ext_vector_type(2)))  float        v2f;
typedef __attribute__((ext_vector_type(8)))  unsigned int v8u;

// fp32 -> bf16 with round-to-nearest-even
__device__ __forceinline__ unsigned short f2bf(float f) {
  unsigned int u = __float_as_uint(f);
  u = (u + 0x7FFFu + ((u >> 16) & 1u)) >> 16;
  return (unsigned short)u;
}

// ---------------- K1: weights fp32 -> bf16 (W1 | W2 | W3 packed) -------------
__global__ void cvt_weights(const float* __restrict__ W1,
                            const float* __restrict__ W2,
                            const float* __restrict__ W3,
                            unsigned short* __restrict__ ws) {
  int i = blockIdx.x * 256 + threadIdx.x;
  if (i < 8192)            ws[i] = f2bf(W1[i]);
  else if (i < 24576)      ws[i] = f2bf(W2[i - 8192]);
  else if (i < 40960)      ws[i] = f2bf(W3[i - 24576]);
}

// ---------------- K2: gather new_xyz + init new_points to -FLT_MAX -----------
__global__ void gather_init(const float* __restrict__ xyz,
                            const int* __restrict__ fps,
                            float* __restrict__ out) {
  int t = blockIdx.x * 256 + threadIdx.x;              // 0 .. 24575
  if (t < B_ * 3 * S_) {
    int b = t / (3 * S_);
    int r = t - b * 3 * S_;
    int c = r / S_;
    int s = r - c * S_;
    out[t] = xyz[(size_t)b * 3 * N_ + (size_t)c * N_ + fps[s]];
  }
  if (t < B_ * O_) out[B_ * 3 * S_ + t] = NEGF;
}

// ---------------- K3: ball-query mask via rank-4 WMMA --------------------------
// d2 < r^2  <=>  s.p - 0.5*|s|^2 > 0.5*(|p|^2 - r^2)
// A (16 samples x K=4) = [sx, sy, sz, -0.5|s|^2], B (4 x 16 points) = [px,py,pz,1]
// D[s][n] = biased dot; mask[n] = (max_s D[s][n]) > rhs[n]
__global__ void __launch_bounds__(256)
ball_mask(const float* __restrict__ xyz,
          const float* __restrict__ newxyz,   // d_out region
          unsigned char* __restrict__ mask) {
  __shared__ __align__(16) float s4[S_ * 4];  // {x, y, z, -0.5*ss} per sample
  int b   = blockIdx.y;
  int tid = threadIdx.x;
  for (int s = tid; s < S_; s += 256) {
    float sx = newxyz[(size_t)b * 3 * S_ + s];
    float sy = newxyz[(size_t)b * 3 * S_ + S_ + s];
    float sz = newxyz[(size_t)b * 3 * S_ + 2 * S_ + s];
    float4 q;
    q.x = sx; q.y = sy; q.z = sz;
    q.w = -0.5f * (sx * sx + sy * sy + sz * sz);
    *(float4*)(s4 + 4 * s) = q;
  }
  __syncthreads();

  int wv    = tid >> 5;
  int l     = tid & 31;
  int nlane = l & 15;
  int half  = l >> 4;
  int n     = blockIdx.x * 128 + wv * 16 + nlane;

  float px = xyz[(size_t)b * 3 * N_ + n];
  float py = xyz[(size_t)b * 3 * N_ + N_ + n];
  float pz = xyz[(size_t)b * 3 * N_ + 2 * N_ + n];
  float rhs = 0.5f * (px * px + py * py + pz * pz - R2_);

#if __has_builtin(__builtin_amdgcn_wmma_f32_16x16x4_f32)
  // B fragment (4x16 f32): VGPR v, half h -> K = v + 2h, N = lane%16.
  // Loop-invariant, built in registers: half0 = {px,py}, half1 = {pz,1}.
  v2f bv;
  bv[0] = half ? pz : px;
  bv[1] = half ? 1.0f : py;

  v8f acc;
#pragma unroll
  for (int v = 0; v < 8; ++v) acc[v] = NEGF;
  v8f z = {};

#pragma unroll 4
  for (int t = 0; t < 64; ++t) {
    // A fragment (16x4 f32): lane row M = lane%16, K pair = {2h, 2h+1}
    v2f a2 = *(const v2f*)(s4 + (t * 16 + nlane) * 4 + 2 * half);
    v8f d = __builtin_amdgcn_wmma_f32_16x16x4_f32(
        false, a2, false, bv, (short)0, z, false, false);
    // D rows are samples: row-max accumulates max over s directly
#pragma unroll
    for (int v = 0; v < 8; ++v) acc[v] = fmaxf(acc[v], d[v]);
  }
  float m8 = acc[0];
#pragma unroll
  for (int v = 1; v < 8; ++v) m8 = fmaxf(m8, acc[v]);
  m8 = fmaxf(m8, __shfl_xor(m8, 16, 32));   // combine the two halves' sample rows
  if (l < 16) mask[(size_t)b * N_ + n] = (m8 > rhs) ? 1 : 0;
#else
  float best = NEGF;
  for (int q = 0; q < S_; ++q) {
    const float* sp = s4 + 4 * q;
    float t = fmaf(sp[0], px, fmaf(sp[1], py, fmaf(sp[2], pz, sp[3])));
    best = fmaxf(best, t);
  }
  mask[(size_t)b * N_ + n] = (best > rhs) ? 1 : 0;
#endif
}

// --------- WMMA fragment helpers (bf16 16x16x32, wave32, contiguous k = 16*half + s) ----------
__device__ __forceinline__ v16bf loadA(const unsigned short* __restrict__ W,
                                       int K, int row, int k0, int half) {
  v8u u;
#pragma unroll
  for (int i = 0; i < 8; ++i) {
    int koff = 16 * half + 2 * i;
    u[i] = *(const unsigned int*)(W + (size_t)row * K + k0 + koff);
  }
  return __builtin_bit_cast(v16bf, u);
}

// B fragment = 16 contiguous K values of one column (column-major LDS)
__device__ __forceinline__ v16bf loadBpk(const unsigned short* colBase, int kbase) {
  const uint4* p = (const uint4*)(colBase + kbase);
  uint4 a = p[0];
  uint4 b = p[1];
  v8u u;
  u[0] = a.x; u[1] = a.y; u[2] = a.z; u[3] = a.w;
  u[4] = b.x; u[5] = b.y; u[6] = b.z; u[7] = b.w;
  return __builtin_bit_cast(v16bf, u);
}

// D rows (m*16+8*half+v) are 8 contiguous rows of one column -> 1 ds_store_b128
__device__ __forceinline__ void storeH(unsigned short* colBase, int m, int half, v8f c) {
  unsigned int d0 = (unsigned int)f2bf(fmaxf(c[0], 0.0f)) |
                    ((unsigned int)f2bf(fmaxf(c[1], 0.0f)) << 16);
  unsigned int d1 = (unsigned int)f2bf(fmaxf(c[2], 0.0f)) |
                    ((unsigned int)f2bf(fmaxf(c[3], 0.0f)) << 16);
  unsigned int d2 = (unsigned int)f2bf(fmaxf(c[4], 0.0f)) |
                    ((unsigned int)f2bf(fmaxf(c[5], 0.0f)) << 16);
  unsigned int d3 = (unsigned int)f2bf(fmaxf(c[6], 0.0f)) |
                    ((unsigned int)f2bf(fmaxf(c[7], 0.0f)) << 16);
  uint4 q; q.x = d0; q.y = d1; q.z = d2; q.w = d3;
  *(uint4*)(colBase + m * 16 + 8 * half) = q;
}

// ---------------- K4: 3-layer MLP via WMMA, 2 column-blocks per wave ---------
// Each A-fragment (weight tile) feeds two WMMAs -> weight traffic halved.
// LDS map (53760 B):
//   [0,18432)      sP : 128 cols x 72 bf16 (column-major P tile, padded)
//   [18432,53248)  per-wave (8704 B): single H buffer, 32 cols x 136 bf16,
//                  overwritten in place layer-to-layer (frags pre-loaded to
//                  regs; same-wave LDS is in-order), then reused as f32
//                  scratch 16 cols x 136 for the max-reduction
//   [53248,53760)  sRed : int[128]
__global__ void __launch_bounds__(128)
mlp_max(const float* __restrict__ points,
        const unsigned short* __restrict__ Wbf,
        const float* __restrict__ b1,
        const float* __restrict__ b2,
        const float* __restrict__ b3,
        const unsigned char* __restrict__ maskp,
        int* __restrict__ outP) {
  __shared__ __align__(16) unsigned char smem[53760];
  unsigned short* sP   = (unsigned short*)smem;
  int*            sRed = (int*)(smem + 53248);

  const unsigned short* W1 = Wbf;
  const unsigned short* W2 = Wbf + 8192;
  const unsigned short* W3 = Wbf + 24576;

  int b   = blockIdx.y;
  int n0  = blockIdx.x * 128;
  int tid = threadIdx.x;
  int w     = tid >> 5;
  int l     = tid & 31;
  int nlane = l & 15;
  int half  = l >> 4;
  int kh    = 16 * half;

  unsigned char*  wreg = smem + 18432 + w * 8704;
  unsigned short* sH   = (unsigned short*)wreg;
  float*          fs   = (float*)wreg;

  sRed[tid] = 0xFF7FFFFF;   // int encoding of -FLT_MAX

  // stage P tile transposed into column-major LDS (float4 global loads)
  {
    for (int e = tid; e < (C_ * 128) / 4; e += 128) {
      int c  = e >> 5;                 // row 0..63
      int jq = e & 31;                 // float4 index 0..31
      float4 f = ((const float4*)(points + ((size_t)b * C_ + c) * N_ + n0))[jq];
      int j = jq * 4;
      sP[(j + 0) * 72 + c] = f2bf(f.x);
      sP[(j + 1) * 72 + c] = f2bf(f.y);
      sP[(j + 2) * 72 + c] = f2bf(f.z);
      sP[(j + 3) * 72 + c] = f2bf(f.w);
    }
  }
  __syncthreads();

  int gn0 = n0 + w * 32 + nlane;           // col-block 0 global n
  bool valid0 = maskp[(size_t)b * N_ + gn0] != 0;
  bool valid1 = maskp[(size_t)b * N_ + gn0 + 16] != 0;

  const unsigned short* colP0 = sP + (w * 32 + nlane) * 72;
  const unsigned short* colP1 = colP0 + 16 * 72;
  unsigned short* colH0 = sH + nlane * 136;
  unsigned short* colH1 = sH + (16 + nlane) * 136;

  // ---- Layer 1: H1 = relu(W1(128x64) @ P(64x32) + b1) ----
  v16bf p0a = loadBpk(colP0, 0 + kh);
  v16bf p1a = loadBpk(colP0, 32 + kh);
  v16bf p0b = loadBpk(colP1, 0 + kh);
  v16bf p1b = loadBpk(colP1, 32 + kh);
#pragma unroll
  for (int m = 0; m < 8; ++m) {
    v8f c0;
#pragma unroll
    for (int v = 0; v < 8; ++v) c0[v] = b1[m * 16 + v + 8 * half];
    v8f c1 = c0;
    v16bf a0 = loadA(W1, 64, m * 16 + nlane, 0, half);
    c0 = __builtin_amdgcn_wmma_f32_16x16x32_bf16(false, a0, false, p0a, (short)0, c0, false, false);
    c1 = __builtin_amdgcn_wmma_f32_16x16x32_bf16(false, a0, false, p0b, (short)0, c1, false, false);
    v16bf a1 = loadA(W1, 64, m * 16 + nlane, 32, half);
    c0 = __builtin_amdgcn_wmma_f32_16x16x32_bf16(false, a1, false, p1a, (short)0, c0, false, false);
    c1 = __builtin_amdgcn_wmma_f32_16x16x32_bf16(false, a1, false, p1b, (short)0, c1, false, false);
    storeH(colH0, m, half, c0);
    storeH(colH1, m, half, c1);
  }
  asm volatile("" ::: "memory");

  // ---- Layer 2: in-place (all B frags register-resident before stores) ----
  v16bf h0a = loadBpk(colH0, 0 + kh);
  v16bf h1a = loadBpk(colH0, 32 + kh);
  v16bf h2a = loadBpk(colH0, 64 + kh);
  v16bf h3a = loadBpk(colH0, 96 + kh);
  v16bf h0b = loadBpk(colH1, 0 + kh);
  v16bf h1b = loadBpk(colH1, 32 + kh);
  v16bf h2b = loadBpk(colH1, 64 + kh);
  v16bf h3b = loadBpk(colH1, 96 + kh);
  asm volatile("" ::: "memory");
#pragma unroll
  for (int m = 0; m < 8; ++m) {
    v8f c0;
#pragma unroll
    for (int v = 0; v < 8; ++v) c0[v] = b2[m * 16 + v + 8 * half];
    v8f c1 = c0;
    v16bf a0 = loadA(W2, 128, m * 16 + nlane, 0, half);
    c0 = __builtin_amdgcn_wmma_f32_16x16x32_bf16(false, a0, false, h0a, (short)0, c0, false, false);
    c1 = __builtin_amdgcn_wmma_f32_16x16x32_bf16(false, a0, false, h0b, (short)0, c1, false, false);
    v16bf a1 = loadA(W2, 128, m * 16 + nlane, 32, half);
    c0 = __builtin_amdgcn_wmma_f32_16x16x32_bf16(false, a1, false, h1a, (short)0, c0, false, false);
    c1 = __builtin_amdgcn_wmma_f32_16x16x32_bf16(false, a1, false, h1b, (short)0, c1, false, false);
    v16bf a2 = loadA(W2, 128, m * 16 + nlane, 64, half);
    c0 = __builtin_amdgcn_wmma_f32_16x16x32_bf16(false, a2, false, h2a, (short)0, c0, false, false);
    c1 = __builtin_amdgcn_wmma_f32_16x16x32_bf16(false, a2, false, h2b, (short)0, c1, false, false);
    v16bf a3 = loadA(W2, 128, m * 16 + nlane, 96, half);
    c0 = __builtin_amdgcn_wmma_f32_16x16x32_bf16(false, a3, false, h3a, (short)0, c0, false, false);
    c1 = __builtin_amdgcn_wmma_f32_16x16x32_bf16(false, a3, false, h3b, (short)0, c1, false, false);
    storeH(colH0, m, half, c0);
    storeH(colH1, m, half, c1);
  }
  asm volatile("" ::: "memory");

  // ---- Layer 3: compute, mask both col-blocks, fmax-combine, f32 scratch ----
  v16bf g0a = loadBpk(colH0, 0 + kh);
  v16bf g1a = loadBpk(colH0, 32 + kh);
  v16bf g2a = loadBpk(colH0, 64 + kh);
  v16bf g3a = loadBpk(colH0, 96 + kh);
  v16bf g0b = loadBpk(colH1, 0 + kh);
  v16bf g1b = loadBpk(colH1, 32 + kh);
  v16bf g2b = loadBpk(colH1, 64 + kh);
  v16bf g3b = loadBpk(colH1, 96 + kh);
  asm volatile("" ::: "memory");

  float* colF = fs + nlane * 136;
#pragma unroll
  for (int m = 0; m < 8; ++m) {
    v8f c0;
#pragma unroll
    for (int v = 0; v < 8; ++v) c0[v] = b3[m * 16 + v + 8 * half];
    v8f c1 = c0;
    v16bf a0 = loadA(W3, 128, m * 16 + nlane, 0, half);
    c0 = __builtin_amdgcn_wmma_f32_16x16x32_bf16(false, a0, false, g0a, (short)0, c0, false, false);
    c1 = __builtin_amdgcn_wmma_f32_16x16x32_bf16(false, a0, false, g0b, (short)0, c1, false, false);
    v16bf a1 = loadA(W3, 128, m * 16 + nlane, 32, half);
    c0 = __builtin_amdgcn_wmma_f32_16x16x32_bf16(false, a1, false, g1a, (short)0, c0, false, false);
    c1 = __builtin_amdgcn_wmma_f32_16x16x32_bf16(false, a1, false, g1b, (short)0, c1, false, false);
    v16bf a2 = loadA(W3, 128, m * 16 + nlane, 64, half);
    c0 = __builtin_amdgcn_wmma_f32_16x16x32_bf16(false, a2, false, g2a, (short)0, c0, false, false);
    c1 = __builtin_amdgcn_wmma_f32_16x16x32_bf16(false, a2, false, g2b, (short)0, c1, false, false);
    v16bf a3 = loadA(W3, 128, m * 16 + nlane, 96, half);
    c0 = __builtin_amdgcn_wmma_f32_16x16x32_bf16(false, a3, false, g3a, (short)0, c0, false, false);
    c1 = __builtin_amdgcn_wmma_f32_16x16x32_bf16(false, a3, false, g3b, (short)0, c1, false, false);
    float4 q0, q1;
    q0.x = fmaxf(valid0 ? fmaxf(c0[0], 0.0f) : NEGF, valid1 ? fmaxf(c1[0], 0.0f) : NEGF);
    q0.y = fmaxf(valid0 ? fmaxf(c0[1], 0.0f) : NEGF, valid1 ? fmaxf(c1[1], 0.0f) : NEGF);
    q0.z = fmaxf(valid0 ? fmaxf(c0[2], 0.0f) : NEGF, valid1 ? fmaxf(c1[2], 0.0f) : NEGF);
    q0.w = fmaxf(valid0 ? fmaxf(c0[3], 0.0f) : NEGF, valid1 ? fmaxf(c1[3], 0.0f) : NEGF);
    q1.x = fmaxf(valid0 ? fmaxf(c0[4], 0.0f) : NEGF, valid1 ? fmaxf(c1[4], 0.0f) : NEGF);
    q1.y = fmaxf(valid0 ? fmaxf(c0[5], 0.0f) : NEGF, valid1 ? fmaxf(c1[5], 0.0f) : NEGF);
    q1.z = fmaxf(valid0 ? fmaxf(c0[6], 0.0f) : NEGF, valid1 ? fmaxf(c1[6], 0.0f) : NEGF);
    q1.w = fmaxf(valid0 ? fmaxf(c0[7], 0.0f) : NEGF, valid1 ? fmaxf(c1[7], 0.0f) : NEGF);
    *(float4*)(colF + m * 16 + 8 * half)     = q0;
    *(float4*)(colF + m * 16 + 8 * half + 4) = q1;
  }
  asm volatile("" ::: "memory");

  // each lane reduces 4 complete rows over this wave's 16 combined columns
#pragma unroll
  for (int rr = 0; rr < 4; ++rr) {
    int row = l * 4 + rr;
    float mv = fs[row];
#pragma unroll
    for (int c2 = 1; c2 < 16; ++c2)
      mv = fmaxf(mv, fs[c2 * 136 + row]);
    atomicMax(&sRed[row], __float_as_int(mv));
  }
  __syncthreads();
  // candidates are -FLT_MAX (negative int encoding) or >= 0 floats, so
  // signed-int max ordering == float max ordering.
  atomicMax(&outP[b * O_ + tid], sRed[tid]);
}

// ---------------------------------------------------------------------------
extern "C" void kernel_launch(void* const* d_in, const int* in_sizes, int n_in,
                              void* d_out, int out_size, void* d_ws, size_t ws_size,
                              hipStream_t stream) {
  const float* xyz    = (const float*)d_in[0];
  const float* points = (const float*)d_in[1];
  const int*   fps    = (const int*)d_in[2];
  const float* W1     = (const float*)d_in[3];
  const float* b1     = (const float*)d_in[4];
  const float* W2     = (const float*)d_in[5];
  const float* b2     = (const float*)d_in[6];
  const float* W3     = (const float*)d_in[7];
  const float* b3     = (const float*)d_in[8];

  float* out = (float*)d_out;                 // [new_xyz 24576 | new_points 1024]
  unsigned short* wbf  = (unsigned short*)d_ws;            // 81920 bytes
  unsigned char*  mask = (unsigned char*)d_ws + 81920;     // 262144 bytes

  cvt_weights<<<160, 256, 0, stream>>>(W1, W2, W3, wbf);
  gather_init<<<96, 256, 0, stream>>>(xyz, fps, out);
  ball_mask<<<dim3(N_ / 128, B_), 256, 0, stream>>>(xyz, out, mask);
  mlp_max<<<dim3(N_ / 128, B_), 128, 0, stream>>>(points, wbf, b1, b2, b3, mask,
                                                  (int*)(out + B_ * 3 * S_));
}
